// CausalSelfAttention_3384434229968
// MI455X (gfx1250) — matrix operand order repaired and checked by reading, not compiled, not run on hardware
//
#include <hip/hip_runtime.h>
#include <hip/hip_bf16.h>

// ---------------------------------------------------------------------------
// Causal self-attention (GPT-2 block) for MI455X / gfx1250, wave32 + WMMA.
// B=4, T=2048, C=768, H=12, hd=64.
// ---------------------------------------------------------------------------

typedef __attribute__((ext_vector_type(16))) _Float16 v16h;
typedef __attribute__((ext_vector_type(8)))  _Float16 v8h;
typedef __attribute__((ext_vector_type(8)))  float    v8f;
typedef int v4i __attribute__((vector_size(4 * sizeof(int))));

#define BB 4
#define TT 2048
#define CC 768
#define HH 12
#define HD 64
#define M_TOT (BB * TT)        // 8192 rows

#if defined(__has_builtin)
#if __has_builtin(__builtin_amdgcn_global_load_async_to_lds_b128)
#define HAVE_ASYNC_LDS 1
#endif
#endif
#ifndef HAVE_ASYNC_LDS
#define HAVE_ASYNC_LDS 0
#endif

#if HAVE_ASYNC_LDS
typedef __attribute__((address_space(1))) v4i* gv4i_ptr;   // global
typedef __attribute__((address_space(3))) v4i* lv4i_ptr;   // LDS
#endif

__device__ __forceinline__ v8f wmma16(v16h a, v16h b, v8f c) {
  return __builtin_amdgcn_wmma_f32_16x16x32_f16(
      /*neg_a=*/false, a, /*neg_b=*/false, b,
      /*c_mod=*/(short)0, c, /*reuse_a=*/false, /*reuse_b=*/false);
}

// A-fragment (16x32 f16, M x K) from row-major src; base = &src[row0*ld + k0].
// ISA layout: lane L -> M = L&15, half = L>>4; elem e<8 -> K=half*8+e,
// e>=8 -> K=16+half*8+(e-8).  Two contiguous 16-byte loads per lane.
__device__ __forceinline__ v16h load_afrag(const _Float16* base, int ld) {
  int lane = threadIdx.x & 31;
  int half = lane >> 4;
  int m    = lane & 15;
  const _Float16* p = base + m * ld;
  v8h lo = *(const v8h*)(p + half * 8);
  v8h hi = *(const v8h*)(p + 16 + half * 8);
  v16h a;
#pragma unroll
  for (int i = 0; i < 8; ++i) { a[i] = lo[i]; a[i + 8] = hi[i]; }
  return a;
}

// B-fragment (32x16 f16, K x N) from a TRANSPOSED matrix bt[N][K] row-major;
// base = &bt[n0*ld + k0].  Lane L -> N = L&15, half = L>>4; elem e<8 -> K =
// half*8 + e, e>=8 -> K=16+half*8+(e-8).  Two 16-byte reads per lane.
__device__ __forceinline__ v16h load_bfrag_t(const _Float16* base, int ld) {
  int lane = threadIdx.x & 31;
  int half = lane >> 4;
  int n    = lane & 15;
  const _Float16* p = base + n * ld;
  v8h lo = *(const v8h*)(p + half * 8);
  v8h hi = *(const v8h*)(p + 16 + half * 8);
  v16h b;
#pragma unroll
  for (int i = 0; i < 8; ++i) { b[i] = lo[i]; b[i + 8] = hi[i]; }
  return b;
}

// ---------------------------------------------------------------------------
// Stage 0: dtype conversion / weight transpose
// ---------------------------------------------------------------------------
__global__ void cvt_f32_to_f16(const float* __restrict__ in,
                               _Float16* __restrict__ out, int n) {
  int i = blockIdx.x * blockDim.x + threadIdx.x;
  if (i < n) out[i] = (_Float16)in[i];
}

// in[K][N] f32 -> out[N][K] f16
__global__ void transpose_to_f16(const float* __restrict__ in,
                                 _Float16* __restrict__ out, int K, int N) {
  int i = blockIdx.x * blockDim.x + threadIdx.x;
  if (i < K * N) {
    int k = i / N, n = i - k * N;
    out[n * K + k] = (_Float16)in[i];
  }
}

// ---------------------------------------------------------------------------
// Stage 1: qkv = x @ w_attn + b_attn, scattered into per-head Q/K/V buffers.
// One wave computes a 16x64 output strip (A-frag reused by 4 WMMAs per
// K-step; 4 independent accumulator chains hide WMMA latency).
// Q gets the 1/sqrt(hd)=0.125 scale folded in; V is stored transposed [hd][T].
// ---------------------------------------------------------------------------
__global__ void qkv_gemm(const _Float16* __restrict__ xh,
                         const _Float16* __restrict__ wt,   // [3C][C] (W^T)
                         const float* __restrict__ bias,    // [3C]
                         _Float16* __restrict__ qh,         // [B,H,T,hd]
                         _Float16* __restrict__ kh,         // [B,H,T,hd]
                         _Float16* __restrict__ vt) {       // [B,H,hd,T]
  const int NT4 = (3 * CC) / 64;  // 36 strips along N
  int wave = blockIdx.x * (blockDim.x >> 5) + (threadIdx.x >> 5);
  int mt = wave / NT4, nt4 = wave - mt * NT4;
  int m0 = mt * 16, n0 = nt4 * 64;

  v8f acc[4];
#pragma unroll
  for (int s = 0; s < 4; ++s) acc[s] = (v8f){0.f,0.f,0.f,0.f,0.f,0.f,0.f,0.f};

#pragma unroll 2
  for (int k0 = 0; k0 < CC; k0 += 32) {
    __builtin_prefetch(xh + m0 * CC + k0 + 64, 0, 1);
    v16h a = load_afrag(xh + m0 * CC + k0, CC);
#pragma unroll
    for (int s = 0; s < 4; ++s) {
      v16h b = load_bfrag_t(wt + (n0 + s * 16) * CC + k0, CC);
      acc[s] = wmma16(a, b, acc[s]);
    }
  }

  int lane = threadIdx.x & 31, half = lane >> 4, nn = lane & 15;
#pragma unroll
  for (int s = 0; s < 4; ++s) {
    int col = n0 + s * 16 + nn;        // [0, 2304)
    float bv = bias[col];
    int which = col / CC;              // 0=q 1=k 2=v
    int c = col - which * CC;
    int h = c >> 6, d = c & 63;
#pragma unroll
    for (int r = 0; r < 8; ++r) {
      int row = m0 + r + 8 * half;     // [0, 8192)
      int b_ = row >> 11, t = row & (TT - 1);
      float val = acc[s][r] + bv;
      size_t bh = (size_t)(b_ * HH + h);
      if (which == 0)
        qh[(bh * TT + t) * HD + d] = (_Float16)(val * 0.125f);
      else if (which == 1)
        kh[(bh * TT + t) * HD + d] = (_Float16)val;
      else
        vt[(bh * HD + d) * TT + t] = (_Float16)val;
    }
  }
}

// ---------------------------------------------------------------------------
// Stage 2: flash-style causal attention.  One wave per (b,h,qtile), 4 waves
// per block.  Keys stream in chunks of 32; K/V chunks are staged into LDS
// with double-buffered GLOBAL_LOAD_ASYNC_TO_LDS_B128 (fallback: ds stores).
// ---------------------------------------------------------------------------

// Stage one 32-key chunk: K rows [key0,key0+32) x 64d (contiguous 4KB) into
// ldsK[32][64], and V^T rows d=0..63 x keys [key0,key0+32) into ldsV[64][32].
__device__ __forceinline__ void stage_kv(_Float16* ldsK, _Float16* ldsV,
                                         const _Float16* K, const _Float16* V,
                                         int key0, int lane) {
  const _Float16* gk = K + (size_t)key0 * HD;   // contiguous 32*64 f16
#if HAVE_ASYNC_LDS
#pragma unroll
  for (int it = 0; it < 8; ++it) {
    __builtin_amdgcn_global_load_async_to_lds_b128(
        (gv4i_ptr)(gk + (it * 32 + lane) * 8),
        (lv4i_ptr)(ldsK + (it * 32 + lane) * 8),
        0, 0);
  }
#pragma unroll
  for (int it = 0; it < 8; ++it) {
    int r  = it * 8 + (lane >> 2);      // d row 0..63
    int c8 = (lane & 3) * 8;            // key sub-offset
    __builtin_amdgcn_global_load_async_to_lds_b128(
        (gv4i_ptr)(V + (size_t)r * TT + key0 + c8),
        (lv4i_ptr)(ldsV + r * 32 + c8),
        0, 0);
  }
#else
#pragma unroll
  for (int it = 0; it < 8; ++it) {
    v8h t = *(const v8h*)(gk + (it * 32 + lane) * 8);
    *(v8h*)(ldsK + (it * 32 + lane) * 8) = t;
  }
#pragma unroll
  for (int it = 0; it < 8; ++it) {
    int r  = it * 8 + (lane >> 2);
    int c8 = (lane & 3) * 8;
    v8h t = *(const v8h*)(V + (size_t)r * TT + key0 + c8);
    *(v8h*)(ldsV + r * 32 + c8) = t;
  }
#endif
}

__device__ __forceinline__ void wait_stage(bool more_in_flight) {
#if HAVE_ASYNC_LDS
  if (more_in_flight) asm volatile("s_wait_asynccnt 0x10" ::: "memory");
  else                asm volatile("s_wait_asynccnt 0x0"  ::: "memory");
#else
  asm volatile("s_wait_dscnt 0x0" ::: "memory");
#endif
}

__global__ void attn_kernel(const _Float16* __restrict__ qh,
                            const _Float16* __restrict__ kh,
                            const _Float16* __restrict__ vt,
                            _Float16* __restrict__ att) {   // [B*T][C] f16
  __shared__ _Float16 ktile[4][2][32 * 64];  // per-wave double buffer
  __shared__ _Float16 vtile[4][2][64 * 32];
  __shared__ _Float16 ptile[4][16][40];      // P bounce buffer (16x32, padded)

  int wid = threadIdx.x >> 5;
  int gw = blockIdx.x * 4 + wid;          // 0 .. 6143
  int qt = gw & 127;                      // q tile (T/16 = 128)
  int bh = gw >> 7;                       // 0 .. 47
  const _Float16* Q = qh + (size_t)bh * TT * HD;
  const _Float16* K = kh + (size_t)bh * TT * HD;
  const _Float16* V = vt + (size_t)bh * HD * TT;

  int lane = threadIdx.x & 31, half = lane >> 4, nn = lane & 15;

  // Q tile A-fragments (d = 0..31, 32..63); softmax scale already folded in.
  v16h qa0 = load_afrag(Q + qt * 16 * HD + 0, HD);
  v16h qa1 = load_afrag(Q + qt * 16 * HD + 32, HD);

  v8f o0 = {0,0,0,0,0,0,0,0}, o1 = o0, o2 = o0, o3 = o0;
  float rmax[8], rsum[8];
#pragma unroll
  for (int r = 0; r < 8; ++r) { rmax[r] = -1e30f; rsum[r] = 0.f; }

  const v8f zero = {0,0,0,0,0,0,0,0};
  int q_end = qt * 16 + 15;
  int nchunks = (q_end >> 5) + 1;

  // prologue: stage chunk 0
  stage_kv(&ktile[wid][0][0], &vtile[wid][0][0], K, V, 0, lane);

  for (int kc = 0; kc < nchunks; ++kc) {
    int key0 = kc * 32;
    bool more = (kc + 1) < nchunks;
    if (more)  // overlap next chunk's DMA with this chunk's math
      stage_kv(&ktile[wid][(kc + 1) & 1][0], &vtile[wid][(kc + 1) & 1][0],
               K, V, key0 + 32, lane);
    wait_stage(more);

    const _Float16* lk = &ktile[wid][kc & 1][0];
    const _Float16* lv = &vtile[wid][kc & 1][0];

    float p0[8], p1[8];
    // --- S = Q K^T for two 16-wide key subtiles (from LDS) ---
#pragma unroll
    for (int s = 0; s < 2; ++s) {
      v16h kb0 = load_bfrag_t(lk + (s * 16) * HD + 0, HD);
      v16h kb1 = load_bfrag_t(lk + (s * 16) * HD + 32, HD);
      v8f sf = wmma16(qa1, kb1, wmma16(qa0, kb0, zero));
#pragma unroll
      for (int r = 0; r < 8; ++r) {
        int row = qt * 16 + r + 8 * half;
        int key = key0 + s * 16 + nn;
        float v = (key <= row) ? sf[r] : -1e30f;      // causal mask
        if (s == 0) p0[r] = v; else p1[r] = v;
      }
    }
    // --- online softmax (row = 16 lanes within a half) ---
#pragma unroll
    for (int r = 0; r < 8; ++r) {
      float m = fmaxf(p0[r], p1[r]);
#pragma unroll
      for (int off = 1; off < 16; off <<= 1) m = fmaxf(m, __shfl_xor(m, off, 32));
      float nm = fmaxf(rmax[r], m);
      float corr = __expf(rmax[r] - nm);
      rmax[r] = nm;
      float e0 = __expf(p0[r] - nm);
      float e1 = __expf(p1[r] - nm);
      p0[r] = e0; p1[r] = e1;
      float srow = e0 + e1;
#pragma unroll
      for (int off = 1; off < 16; off <<= 1) srow += __shfl_xor(srow, off, 32);
      rsum[r] = rsum[r] * corr + srow;
      o0[r] *= corr; o1[r] *= corr; o2[r] *= corr; o3[r] *= corr;
    }
    // --- redistribute P from C-layout to A-layout through LDS ---
#pragma unroll
    for (int r = 0; r < 8; ++r) {
      int m = r + 8 * half;
      ptile[wid][m][nn]      = (_Float16)p0[r];
      ptile[wid][m][16 + nn] = (_Float16)p1[r];
    }
    asm volatile("s_wait_dscnt 0x0" ::: "memory");
    v16h pa;
    {
      const _Float16* pp = &ptile[wid][lane & 15][0];
      v8h lo = *(const v8h*)(pp + half * 8);
      v8h hi = *(const v8h*)(pp + 16 + half * 8);
#pragma unroll
      for (int i = 0; i < 8; ++i) { pa[i] = lo[i]; pa[i + 8] = hi[i]; }
    }
    // --- O += P @ V  (V^T tile in LDS: contiguous B-frag reads) ---
    o0 = wmma16(pa, load_bfrag_t(lv + 0  * 32, 32), o0);
    o1 = wmma16(pa, load_bfrag_t(lv + 16 * 32, 32), o1);
    o2 = wmma16(pa, load_bfrag_t(lv + 32 * 32, 32), o2);
    o3 = wmma16(pa, load_bfrag_t(lv + 48 * 32, 32), o3);
  }

  // --- normalize + write [B*T][C] f16 for proj GEMM ---
  int b_ = bh / HH, h = bh - b_ * HH;
#pragma unroll
  for (int r = 0; r < 8; ++r) {
    float inv = 1.0f / rsum[r];
    int row = qt * 16 + r + 8 * half;
    size_t base = ((size_t)b_ * TT + row) * CC + h * HD;
    att[base + 0  + nn] = (_Float16)(o0[r] * inv);
    att[base + 16 + nn] = (_Float16)(o1[r] * inv);
    att[base + 32 + nn] = (_Float16)(o2[r] * inv);
    att[base + 48 + nn] = (_Float16)(o3[r] * inv);
  }
}

// ---------------------------------------------------------------------------
// Stage 3: out = att @ w_proj + b_proj  -> fp32 d_out (16x64 strip per wave)
// ---------------------------------------------------------------------------
__global__ void proj_gemm(const _Float16* __restrict__ ah,
                          const _Float16* __restrict__ wt,  // [C][C] (W^T)
                          const float* __restrict__ bias,
                          float* __restrict__ out) {
  const int NT4 = CC / 64;  // 12 strips
  int wave = blockIdx.x * (blockDim.x >> 5) + (threadIdx.x >> 5);
  int mt = wave / NT4, nt4 = wave - mt * NT4;
  int m0 = mt * 16, n0 = nt4 * 64;

  v8f acc[4];
#pragma unroll
  for (int s = 0; s < 4; ++s) acc[s] = (v8f){0.f,0.f,0.f,0.f,0.f,0.f,0.f,0.f};

#pragma unroll 2
  for (int k0 = 0; k0 < CC; k0 += 32) {
    __builtin_prefetch(ah + m0 * CC + k0 + 64, 0, 1);
    v16h a = load_afrag(ah + m0 * CC + k0, CC);
#pragma unroll
    for (int s = 0; s < 4; ++s) {
      v16h b = load_bfrag_t(wt + (n0 + s * 16) * CC + k0, CC);
      acc[s] = wmma16(a, b, acc[s]);
    }
  }

  int lane = threadIdx.x & 31, half = lane >> 4, nn = lane & 15;
#pragma unroll
  for (int s = 0; s < 4; ++s) {
    int col = n0 + s * 16 + nn;
    float bv = bias[col];
#pragma unroll
    for (int r = 0; r < 8; ++r) {
      int row = m0 + r + 8 * half;
      out[(size_t)row * CC + col] = acc[s][r] + bv;
    }
  }
}

// ---------------------------------------------------------------------------
// Host-side launch
// ---------------------------------------------------------------------------
extern "C" void kernel_launch(void* const* d_in, const int* in_sizes, int n_in,
                              void* d_out, int out_size, void* d_ws, size_t ws_size,
                              hipStream_t stream) {
  const float* x      = (const float*)d_in[0];   // [B,T,C]
  const float* w_attn = (const float*)d_in[1];   // [C,3C]
  const float* b_attn = (const float*)d_in[2];   // [3C]
  const float* w_proj = (const float*)d_in[3];   // [C,C]
  const float* b_proj = (const float*)d_in[4];   // [C]
  float* out = (float*)d_out;

  // workspace carve-up (f16 buffers), 256-B aligned
  char* ws = (char*)d_ws;
  size_t off = 0;
  auto carve = [&](size_t bytes) {
    char* p = ws + off;
    off = (off + bytes + 255) & ~(size_t)255;
    return p;
  };
  _Float16* xh      = (_Float16*)carve((size_t)M_TOT * CC * 2);       // 12.6 MB
  _Float16* wattn_t = (_Float16*)carve((size_t)3 * CC * CC * 2);      //  3.5 MB
  _Float16* wproj_t = (_Float16*)carve((size_t)CC * CC * 2);          //  1.2 MB
  _Float16* qh      = (_Float16*)carve((size_t)M_TOT * CC * 2);
  _Float16* kh      = (_Float16*)carve((size_t)M_TOT * CC * 2);
  _Float16* vt      = (_Float16*)carve((size_t)M_TOT * CC * 2);
  _Float16* att     = (_Float16*)carve((size_t)M_TOT * CC * 2);

  // Stage 0: conversions
  {
    int n = M_TOT * CC;
    cvt_f32_to_f16<<<(n + 255) / 256, 256, 0, stream>>>(x, xh, n);
    int nw = CC * 3 * CC;
    transpose_to_f16<<<(nw + 255) / 256, 256, 0, stream>>>(w_attn, wattn_t, CC, 3 * CC);
    int np = CC * CC;
    transpose_to_f16<<<(np + 255) / 256, 256, 0, stream>>>(w_proj, wproj_t, CC, CC);
  }

  // Stage 1: QKV GEMM — 512 x 36 strips, 8 waves/block
  {
    int waves = (M_TOT / 16) * ((3 * CC) / 64);   // 18432
    qkv_gemm<<<waves / 8, 256, 0, stream>>>(xh, wattn_t, b_attn, qh, kh, vt);
  }

  // Stage 2: attention — 48 heads * 128 q-tiles = 6144 waves, 4 waves/block
  attn_kernel<<<6144 / 4, 128, 0, stream>>>(qh, kh, vt, att);

  // Stage 3: proj GEMM — 512 x 12 strips, 8 waves/block
  {
    int waves = (M_TOT / 16) * (CC / 64);         // 6144
    proj_gemm<<<waves / 8, 256, 0, stream>>>(att, wproj_t, b_proj, out);
  }
}
